// GNNModel_17480516894920
// MI455X (gfx1250) — compile-verified, matrix operand order checked by reading, not run on hardware
//
#include <hip/hip_runtime.h>
#include <hip/hip_bf16.h>

typedef __attribute__((ext_vector_type(16))) __bf16 v16bf;
typedef __attribute__((ext_vector_type(8)))  __bf16 v8bf;
typedef __attribute__((ext_vector_type(8)))  float  v8f;
typedef __attribute__((ext_vector_type(4)))  int    v4i;

#define BM 128
#define BN 64
#define BK 32
#define LDT (BK + 8)   // 40 bf16 = 80B row stride (16B aligned), A and B-transposed tiles

#if __has_builtin(__builtin_amdgcn_global_load_async_to_lds_b128) && __has_builtin(__builtin_amdgcn_s_wait_asynccnt)
#define USE_ASYNC 1
#else
#define USE_ASYNC 0
#endif

// 16-byte global->LDS chunk copy (async-to-LDS on CDNA5 when available)
__device__ __forceinline__ void copy16(__bf16* l, const __bf16* g) {
#if USE_ASYNC
    __builtin_amdgcn_global_load_async_to_lds_b128(
        (v4i*)g,
        (__attribute__((address_space(3))) v4i*)l, 0, 0);
#else
    *(uint4*)l = *(const uint4*)g;
#endif
}

__device__ __forceinline__ void wait_stage() {
#if USE_ASYNC
    __builtin_amdgcn_s_wait_asynccnt(0);
#endif
}

// two contiguous 16B runs -> one WMMA 16-bit operand fragment
__device__ __forceinline__ v16bf load_frag(const __bf16* base, int kb) {
    v8bf lo = *(const v8bf*)(base + kb);
    v8bf hi = *(const v8bf*)(base + 16 + kb);
    return __builtin_shufflevector(lo, hi, 0, 1, 2, 3, 4, 5, 6, 7,
                                   8, 9, 10, 11, 12, 13, 14, 15);
}

// ---------------------------------------------------------------- utilities

__global__ void gcn_init_ones(float* __restrict__ d, int n) {
    int i = blockIdx.x * blockDim.x + threadIdx.x;
    if (i < n) d[i] = 1.0f;
}

__global__ void gcn_count_deg(float* __restrict__ deg, const int* __restrict__ dst, int e) {
    int i = blockIdx.x * blockDim.x + threadIdx.x;
    if (i < e) unsafeAtomicAdd(&deg[dst[i]], 1.0f);
}

__global__ void gcn_rsqrt_ip(float* __restrict__ d, int n) {
    int i = blockIdx.x * blockDim.x + threadIdx.x;
    if (i < n) d[i] = rsqrtf(d[i]);
}

__global__ void gcn_cvt_bf16(const float* __restrict__ s, __bf16* __restrict__ d, int n) {
    int i = blockIdx.x * blockDim.x + threadIdx.x;
    if (i < n) d[i] = (__bf16)s[i];
}

// W[k][n] (f32) -> WT[n][k] (bf16): one-time transpose so B tiles stage contiguously
__global__ void gcn_cvt_wT(const float* __restrict__ W, __bf16* __restrict__ WT, int K, int N) {
    int i = blockIdx.x * blockDim.x + threadIdx.x;
    if (i >= K * N) return;
    int k = i / N, n = i - k * N;
    WT[(size_t)n * K + k] = (__bf16)W[i];
}

// ------------------------------------------------- fused GEMM + self-loop
// h = Xb(MxK,bf16) * WbT^T(KxN)   (bf16 WMMA, f32 accumulate)
// agg = h * dinv^2                (self-loop init for the edge scatter)
__global__ __launch_bounds__(256)
void gcn_gemm_agg(const __bf16* __restrict__ Xb, const __bf16* __restrict__ WbT,
                  float* __restrict__ h, float* __restrict__ agg,
                  const float* __restrict__ dinv,
                  int M, int K, int Nc) {
    __shared__ alignas(16) __bf16 As[2][BM * LDT];
    __shared__ alignas(16) __bf16 Bs[2][BN * LDT];   // transposed: [n][k]

    const int t    = threadIdx.x;
    const int lane = t & 31;
    const int wave = t >> 5;
    const int wm   = wave & 3;   // 4 wave-rows of 32 (M)
    const int wn   = wave >> 2;  // 2 wave-cols of 32 (N)
    const int mBase = blockIdx.x * BM;
    const int nBase = blockIdx.y * BN;

    // stage one BK-tile into buffer `buf` (pure 16B chunk copies, no divergence)
    auto stage = [&](int k0, int buf) {
#pragma unroll
        for (int j = 0; j < 2; ++j) {               // A: 512 chunks / 256 threads
            int c   = t + j * 256;
            int row = c >> 2;                        // 4 chunks per 32-wide row
            int col = (c & 3) << 3;
            int gr  = mBase + row; if (gr > M - 1) gr = M - 1;
            copy16(&As[buf][row * LDT + col], Xb + (size_t)gr * K + k0 + col);
        }
        {                                            // B: 256 chunks / 256 threads
            int n   = t >> 2;
            int col = (t & 3) << 3;
            int gn  = nBase + n; if (gn > Nc - 1) gn = Nc - 1;
            copy16(&Bs[buf][n * LDT + col], WbT + (size_t)gn * K + k0 + col);
        }
    };

    v8f acc[2][2];
#pragma unroll
    for (int tm = 0; tm < 2; ++tm)
#pragma unroll
        for (int tn = 0; tn < 2; ++tn)
#pragma unroll
            for (int r = 0; r < 8; ++r) acc[tm][tn][r] = 0.0f;

    const int nt = K / BK;
    stage(0, 0);
    wait_stage();
    __syncthreads();

    const int kb   = (lane & 16) ? 8 : 0;           // per-lane K-half select
    const int arow = wm * 32 + (lane & 15);
    const int brow = wn * 32 + (lane & 15);

    for (int kt = 0; kt < nt; ++kt) {
        const int buf = kt & 1;
        if (kt + 1 < nt) stage((kt + 1) * BK, buf ^ 1);

        const __bf16* Ab = As[buf];
        const __bf16* Bb = Bs[buf];
        v16bf a0 = load_frag(Ab + (size_t)arow * LDT, kb);
        v16bf a1 = load_frag(Ab + (size_t)(arow + 16) * LDT, kb);
        v16bf b0 = load_frag(Bb + (size_t)brow * LDT, kb);
        v16bf b1 = load_frag(Bb + (size_t)(brow + 16) * LDT, kb);

        acc[0][0] = __builtin_amdgcn_wmma_f32_16x16x32_bf16(false, a0, false, b0, (short)0, acc[0][0], false, false);
        acc[0][1] = __builtin_amdgcn_wmma_f32_16x16x32_bf16(false, a0, false, b1, (short)0, acc[0][1], false, false);
        acc[1][0] = __builtin_amdgcn_wmma_f32_16x16x32_bf16(false, a1, false, b0, (short)0, acc[1][0], false, false);
        acc[1][1] = __builtin_amdgcn_wmma_f32_16x16x32_bf16(false, a1, false, b1, (short)0, acc[1][1], false, false);

        wait_stage();
        __syncthreads();
    }

    // C/D layout: lane<16 -> rows 0..7, lane>=16 -> rows 8..15; col = lane&15
    const int rhalf = (lane & 16) ? 8 : 0;
    const int ccol  = lane & 15;
#pragma unroll
    for (int tm = 0; tm < 2; ++tm) {
#pragma unroll
        for (int tn = 0; tn < 2; ++tn) {
            const int gn = nBase + wn * 32 + tn * 16 + ccol;
            if (gn >= Nc) continue;
#pragma unroll
            for (int r = 0; r < 8; ++r) {
                const int gm = mBase + wm * 32 + tm * 16 + rhalf + r;
                if (gm >= M) continue;
                const float v  = acc[tm][tn][r];
                const size_t o = (size_t)gm * Nc + gn;
                const float di = dinv[gm];
                h[o]   = v;
                agg[o] = v * di * di;
            }
        }
    }
}

// ------------------------------------------------- edge scatter (fp32 atomics)
__global__ void gcn_scatter(const float* __restrict__ h, float* __restrict__ agg,
                            const int* __restrict__ src, const int* __restrict__ dst,
                            const float* __restrict__ dinv, int e, int Nc) {
    const int chunks = Nc >> 2;
    long tid = (long)blockIdx.x * blockDim.x + threadIdx.x;
    int  ei  = (int)(tid / chunks);
    if (ei >= e) return;
    int  cb  = (int)(tid % chunks) << 2;
    int  s = src[ei], d = dst[ei];
    float coef = dinv[s] * dinv[d];
    float4 hv = *(const float4*)(h + (size_t)s * Nc + cb);
    float* ap = agg + (size_t)d * Nc + cb;
    unsafeAtomicAdd(ap + 0, hv.x * coef);
    unsafeAtomicAdd(ap + 1, hv.y * coef);
    unsafeAtomicAdd(ap + 2, hv.z * coef);
    unsafeAtomicAdd(ap + 3, hv.w * coef);
}

// ------------------------------------------------- bias + skip + relu
__global__ void gcn_finalize_bf16(const float* __restrict__ agg, const float* __restrict__ bias,
                                  const __bf16* __restrict__ skip, __bf16* __restrict__ out,
                                  int total, int Nc) {
    int i = blockIdx.x * blockDim.x + threadIdx.x;
    if (i >= total) return;
    int c = i % Nc;
    float v = agg[i] + bias[c];
    if (skip) v += (float)skip[i];
    out[i] = (__bf16)fmaxf(v, 0.0f);
}

__global__ void gcn_finalize_f32(const float* __restrict__ agg, const float* __restrict__ bias,
                                 float* __restrict__ out, int total, int Nc) {
    int i = blockIdx.x * blockDim.x + threadIdx.x;
    if (i >= total) return;
    int c = i % Nc;
    out[i] = fmaxf(agg[i] + bias[c], 0.0f);
}

// ---------------------------------------------------------------- launch

extern "C" void kernel_launch(void* const* d_in, const int* in_sizes, int n_in,
                              void* d_out, int out_size, void* d_ws, size_t ws_size,
                              hipStream_t stream) {
    (void)in_sizes; (void)n_in; (void)out_size; (void)ws_size;
    const int N = 10000, E = 160000;
    static const int dims[11] = {128, 512, 256, 128, 64, 32, 64, 128, 256, 512, 128};

    const float* x    = (const float*)d_in[0];
    const int*   esrc = (const int*)d_in[1];
    const int*   edst = (const int*)d_in[2];
    const float* W[10]; const float* B[10];
    for (int i = 0; i < 10; ++i) { W[i] = (const float*)d_in[3 + i]; B[i] = (const float*)d_in[13 + i]; }

    char* ws = (char*)d_ws;
    size_t off = 0;
    auto carve = [&](size_t bytes) -> char* {
        char* p = ws + off;
        off = (off + bytes + 255) & ~(size_t)255;
        return p;
    };
    float*  dinv = (float*)carve((size_t)N * sizeof(float));
    float*  h    = (float*)carve((size_t)N * 512 * sizeof(float));
    float*  agg  = (float*)carve((size_t)N * 512 * sizeof(float));
    __bf16* xb   = (__bf16*)carve((size_t)N * 128 * sizeof(__bf16));
    __bf16* WbT[10];
    for (int i = 0; i < 10; ++i) WbT[i] = (__bf16*)carve((size_t)dims[i] * dims[i + 1] * sizeof(__bf16));
    __bf16* act[9];
    for (int i = 0; i < 9; ++i) act[i] = (__bf16*)carve((size_t)N * dims[i + 1] * sizeof(__bf16));

    // degree with self-loop, then dinv = rsqrt(deg)
    gcn_init_ones<<<(N + 255) / 256, 256, 0, stream>>>(dinv, N);
    gcn_count_deg<<<(E + 255) / 256, 256, 0, stream>>>(dinv, edst, E);
    gcn_rsqrt_ip<<<(N + 255) / 256, 256, 0, stream>>>(dinv, N);

    // x -> bf16; W -> bf16 transposed [fout][fin]
    {
        int n = N * 128;
        gcn_cvt_bf16<<<(n + 255) / 256, 256, 0, stream>>>(x, xb, n);
    }
    for (int i = 0; i < 10; ++i) {
        int n = dims[i] * dims[i + 1];
        gcn_cvt_wT<<<(n + 255) / 256, 256, 0, stream>>>(W[i], WbT[i], dims[i], dims[i + 1]);
    }

    for (int i = 0; i < 10; ++i) {
        const int K = dims[i], Nc = dims[i + 1];
        const __bf16* in = (i == 0) ? xb : act[i - 1];
        const __bf16* skip = nullptr;
        if (i == 5) skip = act[3];
        if (i == 6) skip = act[2];
        if (i == 7) skip = act[1];
        if (i == 8) skip = act[0];

        dim3 gg((N + BM - 1) / BM, (Nc + BN - 1) / BN);
        gcn_gemm_agg<<<gg, 256, 0, stream>>>(in, WbT[i], h, agg, dinv, N, K, Nc);

        long total = (long)E * (Nc >> 2);
        gcn_scatter<<<(unsigned)((total + 255) / 256), 256, 0, stream>>>(h, agg, esrc, edst, dinv, E, Nc);

        int tot2 = N * Nc;
        if (i == 9)
            gcn_finalize_f32<<<(tot2 + 255) / 256, 256, 0, stream>>>(agg, B[i], (float*)d_out, tot2, Nc);
        else
            gcn_finalize_bf16<<<(tot2 + 255) / 256, 256, 0, stream>>>(agg, B[i], skip, act[i], tot2, Nc);
    }
}
